// SceneGraphModel_34797825032680
// MI455X (gfx1250) — matrix operand order repaired and checked by reading, not compiled, not run on hardware
//
#include <hip/hip_runtime.h>
#include <hip/hip_bf16.h>

typedef __attribute__((ext_vector_type(16))) _Float16 v16h;
typedef __attribute__((ext_vector_type(8)))  _Float16 v8h;
typedef __attribute__((ext_vector_type(8)))  float    v8f;

#define COUT 24
#define AS3 __attribute__((address_space(3)))

// LDS byte-offset of a generic pointer to a __shared__ object (AS3 ptrs are
// 32-bit LDS offsets on amdgcn).
static __device__ __forceinline__ unsigned lds_offset_of(const void* p) {
  return (unsigned)(unsigned long long)(AS3 const char*)p;
}

// CDNA5 async global->LDS copy, 16 bytes per lane, tracked with ASYNCcnt.
static __device__ __forceinline__ void async_copy_b128(unsigned lds_off,
                                                       const void* gaddr) {
  asm volatile("global_load_async_to_lds_b128 %0, %1, off"
               :: "v"(lds_off), "v"((unsigned long long)gaddr)
               : "memory");
}
#define S_WAIT_ASYNCCNT_0() asm volatile("s_wait_asynccnt 0x0" ::: "memory")
#define S_WAIT_ASYNCCNT_2() asm volatile("s_wait_asynccnt 0x2" ::: "memory")
// Split workgroup barrier WITHOUT the compiler's implicit loadcnt drain:
// cross-wave visibility of the LDS tile is guaranteed by s_wait_asynccnt
// before signaling (producer side) and by the s_wait_dscnt the WMMA operand
// consumption forces (consumer side). Private global loads (A fragments)
// are allowed to remain in flight across the barrier.
#define WG_BARRIER() asm volatile("s_barrier_signal -1\n\ts_barrier_wait -1" ::: "memory")

// ---------------------------------------------------------------------------
// Conv 3x3 stride2 pad1 + bias + relu + BN (fp32, weights staged in LDS)
// ---------------------------------------------------------------------------
__global__ __launch_bounds__(256)
void conv_bn_relu(const float* __restrict__ in,
                  const float* __restrict__ W,
                  const float* __restrict__ bias,
                  const float* __restrict__ gam,
                  const float* __restrict__ bet,
                  const float* __restrict__ mu,
                  const float* __restrict__ var,
                  float* __restrict__ out,
                  int B, int Cin, int Hin, int Win) {
  const int Ho = Hin >> 1, Wo = Win >> 1;
  __shared__ float sW[COUT * 24 * 9];
  __shared__ float sB[COUT], sS[COUT], sT[COUT];
  const int nw = COUT * Cin * 9;
  for (int i = threadIdx.x; i < nw; i += blockDim.x) sW[i] = W[i];
  if (threadIdx.x < COUT) {
    int c = threadIdx.x;
    float s = gam[c] * rsqrtf(var[c] + 1e-5f);
    sB[c] = bias[c]; sS[c] = s; sT[c] = bet[c] - mu[c] * s;
  }
  __syncthreads();
  long long idx = (long long)blockIdx.x * blockDim.x + threadIdx.x;
  long long total = (long long)B * Ho * Wo;
  if (idx >= total) return;
  int wo = (int)(idx % Wo); long long t = idx / Wo;
  int ho = (int)(t % Ho);  int n = (int)(t / Ho);

  float acc[COUT];
#pragma unroll
  for (int c = 0; c < COUT; ++c) acc[c] = 0.f;

  for (int ci = 0; ci < Cin; ++ci) {
    const float* ip = in + ((long long)(n * Cin + ci) * Hin) * Win;
    const float* wci = sW + ci * 9;
#pragma unroll
    for (int kh = 0; kh < 3; ++kh) {
      int hy = 2 * ho - 1 + kh;
      if (hy < 0 || hy >= Hin) continue;
#pragma unroll
      for (int kw = 0; kw < 3; ++kw) {
        int wx = 2 * wo - 1 + kw;
        if (wx < 0 || wx >= Win) continue;
        float v = ip[(long long)hy * Win + wx];
        const float* wp = wci + kh * 3 + kw;
#pragma unroll
        for (int c = 0; c < COUT; ++c) acc[c] += v * wp[(long long)c * Cin * 9];
      }
    }
  }
  long long plane = (long long)Ho * Wo;
  long long obase = (long long)n * COUT * plane + (long long)ho * Wo + wo;
#pragma unroll
  for (int c = 0; c < COUT; ++c) {
    float y = acc[c] + sB[c];
    y = fmaxf(y, 0.f);
    y = y * sS[c] + sT[c];
    out[obase + (long long)c * plane] = y;
  }
}

// ---------------------------------------------------------------------------
// ROI align (fm: B x 24 x 16 x 16) -> obj f16 matrix (1024 x 1184 padded)
// ---------------------------------------------------------------------------
__global__ void roi_align_obj(const float* __restrict__ fm,
                              const float* __restrict__ boxes,
                              const int* __restrict__ bidx,
                              _Float16* __restrict__ obj,
                              int Nb) {
  int idx = blockIdx.x * blockDim.x + threadIdx.x;
  int total = Nb * 24 * 49;
  if (idx >= total) return;
  int px = idx % 7; int t = idx / 7;
  int py = t % 7;  t /= 7;
  int c = t % 24;  int box = t / 24;
  const int H = 16, Wd = 16;
  const float SC = 1.0f / 16.0f;
  float x1 = boxes[box * 4 + 0] * SC - 0.5f;
  float y1 = boxes[box * 4 + 1] * SC - 0.5f;
  float x2 = boxes[box * 4 + 2] * SC - 0.5f;
  float y2 = boxes[box * 4 + 3] * SC - 0.5f;
  float bw = (x2 - x1) / 7.0f, bh = (y2 - y1) / 7.0f;
  float xx = x1 + ((float)px + 0.5f) * bw;
  float yy = y1 + ((float)py + 0.5f) * bh;
  bool valid = (yy > -1.f) && (yy < (float)H) && (xx > -1.f) && (xx < (float)Wd);
  float y = fminf(fmaxf(yy, 0.f), (float)(H - 1));
  float x = fminf(fmaxf(xx, 0.f), (float)(Wd - 1));
  int y0 = (int)floorf(y), x0 = (int)floorf(x);
  int y1i = (y0 + 1 < H) ? y0 + 1 : H - 1;
  int x1i = (x0 + 1 < Wd) ? x0 + 1 : Wd - 1;
  float ly = y - (float)y0, lx = x - (float)x0;
  const float* f = fm + ((long long)bidx[box] * 24 + c) * (H * Wd);
  float v00 = f[y0 * Wd + x0],  v01 = f[y0 * Wd + x1i];
  float v10 = f[y1i * Wd + x0], v11 = f[y1i * Wd + x1i];
  float val = (1.f - ly) * (1.f - lx) * v00 + (1.f - ly) * lx * v01 +
              ly * (1.f - lx) * v10 + ly * lx * v11;
  val = valid ? val : 0.f;
  obj[(long long)box * 1184 + c * 49 + py * 7 + px] = (_Float16)val;
}

__global__ void obj_tail(const float* __restrict__ pred, _Float16* __restrict__ obj, int Nb) {
  int idx = blockIdx.x * blockDim.x + threadIdx.x;
  if (idx >= Nb * 8) return;
  int i = idx & 7, box = idx >> 3;
  float v = (i < 4) ? pred[box * 4 + i] : 0.f;
  obj[(long long)box * 1184 + 1176 + i] = (_Float16)v;
}

// ---------------------------------------------------------------------------
// Weight fp32 -> f16 conversion with K padding
// ---------------------------------------------------------------------------
__global__ void wcvt_pad(const float* __restrict__ W, _Float16* __restrict__ Wh,
                         int N, int K, int Kp) {
  long long idx = (long long)blockIdx.x * blockDim.x + threadIdx.x;
  if (idx >= (long long)N * Kp) return;
  int kk = (int)(idx % Kp); int n = (int)(idx / Kp);
  float v = (kk < K) ? W[(long long)n * K + kk] : 0.f;
  Wh[idx] = (_Float16)v;
}

// relate layer0: K=2360 -> Kp=2368 with 4-zero seam after 1180 so that the
// gathered A rows (two padded 1184-wide obj rows) line up with the weights.
__global__ void wcvt_relate0(const float* __restrict__ W, _Float16* __restrict__ Wh, int N) {
  long long idx = (long long)blockIdx.x * blockDim.x + threadIdx.x;
  if (idx >= (long long)N * 2368) return;
  int kk = (int)(idx % 2368); int n = (int)(idx / 2368);
  float v = 0.f;
  if (kk < 1180)                       v = W[(long long)n * 2360 + kk];
  else if (kk >= 1184 && kk < 2364)    v = W[(long long)n * 2360 + kk - 4];
  Wh[idx] = (_Float16)v;
}

// ---------------------------------------------------------------------------
// WMMA GEMM: H = bn(relu(A @ W^T + b)) in f16.
// Block = 8 waves = 128(M) x 128(N) macro-tile; wave = 16x128 strip: one A
// fragment feeds 8 WMMAs per K-step (halves barrier/async/A-load overhead
// per FLOP vs a 16x64 strip). B tile (128 cols x 32 K = 8 KB) staged in LDS
// via 2 async b128 copies per thread, double-buffered (ASYNCcnt, in-order
// completion => steady-state wait asynccnt<=2), split asm barriers.
// Steady-state K loop is branch-free (last step peeled).
// GATHER: rows are [obj[pairs[r][0]] | obj[pairs[r][1]]], obj stride 1184;
// the 1184 seam is 32-aligned so the base pick is uniform per K-step.
// A/B fragments use contiguous k = 16*hi + s, same order in both (wave32).
// ---------------------------------------------------------------------------
template <bool GATHER>
__global__ __launch_bounds__(256)
void gemm_wmma_bnrelu(const _Float16* __restrict__ A,
                      const int* __restrict__ pairs,
                      const _Float16* __restrict__ W,
                      const float* __restrict__ bias,
                      const float* __restrict__ gam,
                      const float* __restrict__ bet,
                      const float* __restrict__ mu,
                      const float* __restrict__ var,
                      _Float16* __restrict__ H,
                      int M, int Kp, int N) {
  __shared__ _Float16 ldsB[2][128 * 32];  // [buf][col*32 + k]

  const int tid  = threadIdx.x;
  const int lane = tid & 31;
  const int ln   = lane & 15;
  const int hi   = lane >> 4;
  const int wv   = tid >> 5;              // 0..7

  const int tilesN = N >> 7;
  const int tm = (blockIdx.x / tilesN) * 128 + wv * 16;
  const int tn = (blockIdx.x % tilesN) << 7;

  // --- B staging: thread t copies 32B of one column: col = t>>1,
  //     half-chunk = (t&1)*16 halves, as two async b128 copies.
  const int scol   = tid >> 1;
  const int schunk = (tid & 1) * 16;
  const _Float16* gB = W + (size_t)(tn + scol) * Kp + schunk;
  const unsigned myLds0 = lds_offset_of(&ldsB[0][scol * 32 + schunk]);
  const unsigned myLds1 = lds_offset_of(&ldsB[1][scol * 32 + schunk]);

  // --- A row base pointers (gather: uniform p0/p1 switch at k0 == 1184)
  const int rowA = tm + ln;
  const _Float16* base0;
  const _Float16* base1;
  if (GATHER) {
    base0 = A + (size_t)pairs[rowA * 2 + 0] * 1184;
    base1 = A + (size_t)pairs[rowA * 2 + 1] * 1184 - 1184;
  } else {
    base0 = A + (size_t)rowA * Kp;
    base1 = base0;
  }

  union AU { v16h v; v8h h[2]; };
  auto loadA = [&](int k0) -> AU {
    const _Float16* ab = (GATHER && k0 >= 1184) ? base1 : base0;
    AU u;
    int ka = k0 + 16 * hi;    // chunk0: K = k0 + 16*hi .. +7
    u.h[0] = *(const v8h*)(ab + ka);
    u.h[1] = *(const v8h*)(ab + ka + 8);
    return u;
  };

  v8f c[8] = {};
  const int nsteps = Kp >> 5;
  const unsigned bOff = (unsigned)((size_t)ln * 32 + 16 * hi);

  // prologue: async-prefetch B tile 0 (2 copies), preload A fragment 0
  async_copy_b128(myLds0, gB);
  async_copy_b128(myLds0 + 16, (const char*)gB + 16);
  AU areg = loadA(0);

  // steady state: branch-free, double-buffered
  for (int s = 0; s < nsteps - 1; ++s) {
    const unsigned dst = (s & 1) ? myLds0 : myLds1;
    const _Float16* src = gB + ((s + 1) << 5);
    async_copy_b128(dst, src);
    async_copy_b128(dst + 16, (const char*)src + 16);
    S_WAIT_ASYNCCNT_2();       // this step's pair (issued last iter) is done
    WG_BARRIER();              // publish buf[s&1] to all waves

    AU anext = loadA((s + 1) << 5);   // stays in flight across barriers

    const _Float16* bp = &ldsB[s & 1][bOff];
#pragma unroll
    for (int j = 0; j < 8; ++j) {
      v16h b = *(const v16h*)(bp + (size_t)j * 16 * 32);
      c[j] = __builtin_amdgcn_wmma_f32_16x16x32_f16(false, areg.v, false, b,
                                                    (short)0, c[j], false, false);
    }

    WG_BARRIER();              // all waves done reading buf[s&1] (dscnt==0 via wmma use)
    areg = anext;
  }

  // epilogue step (s == nsteps-1)
  {
    S_WAIT_ASYNCCNT_0();
    WG_BARRIER();
    const _Float16* bp = &ldsB[(nsteps - 1) & 1][bOff];
#pragma unroll
    for (int j = 0; j < 8; ++j) {
      v16h b = *(const v16h*)(bp + (size_t)j * 16 * 32);
      c[j] = __builtin_amdgcn_wmma_f32_16x16x32_f16(false, areg.v, false, b,
                                                    (short)0, c[j], false, false);
    }
  }

#pragma unroll
  for (int j = 0; j < 8; ++j) {
    int col = tn + 16 * j + ln;          // constant per lane within c[j]
    float bb = bias[col];
    float s = gam[col] * rsqrtf(var[col] + 1e-5f);
    float t = bet[col] - mu[col] * s;
#pragma unroll
    for (int v = 0; v < 8; ++v) {
      int row = tm + v + 8 * hi;         // C/D layout: VGPR v -> M = v (+8 hi)
      float x = c[j][v] + bb;
      x = fmaxf(x, 0.f);
      x = x * s + t;
      H[(size_t)row * N + col] = (_Float16)x;
    }
  }
}

// ---------------------------------------------------------------------------
// Output layer (dout <= 8) + softmax (act=0) or sigmoid (act=1)
// ---------------------------------------------------------------------------
__global__ void head_out(const _Float16* __restrict__ Hm,
                         const float* __restrict__ Wo,
                         const float* __restrict__ bo,
                         float* __restrict__ out,
                         int M, int K, int dout, int act) {
  int row = blockIdx.x * blockDim.x + threadIdx.x;
  if (row >= M) return;
  float logit[8];
  for (int o = 0; o < dout; ++o) logit[o] = bo[o];
  const _Float16* h = Hm + (size_t)row * K;
  for (int k = 0; k < K; ++k) {
    float hv = (float)h[k];
    for (int o = 0; o < dout; ++o) logit[o] += hv * Wo[(size_t)o * K + k];
  }
  float* orow = out + (size_t)row * dout;
  if (act == 0) {
    float m = logit[0];
    for (int o = 1; o < dout; ++o) m = fmaxf(m, logit[o]);
    float sum = 0.f;
    for (int o = 0; o < dout; ++o) { float e = __expf(logit[o] - m); logit[o] = e; sum += e; }
    float inv = 1.f / sum;
    for (int o = 0; o < dout; ++o) orow[o] = logit[o] * inv;
  } else {
    for (int o = 0; o < dout; ++o) orow[o] = 1.f / (1.f + __expf(-logit[o]));
  }
}

// ---------------------------------------------------------------------------
// Launch
// ---------------------------------------------------------------------------
extern "C" void kernel_launch(void* const* d_in, const int* in_sizes, int n_in,
                              void* d_out, int out_size, void* d_ws, size_t ws_size,
                              hipStream_t stream) {
  (void)in_sizes; (void)n_in; (void)out_size; (void)ws_size;
  const float* features    = (const float*)d_in[0];
  const float* orig_bboxes = (const float*)d_in[1];
  const float* pred_bboxes = (const float*)d_in[2];
  const int*   box_bidx    = (const int*)d_in[3];
  const int*   rela_pairs  = (const int*)d_in[4];
  float* out = (float*)d_out;

  const int B = 64, NB = 1024, NR = 15360, LAT = 1024;
  const int KP_OBJ = 1184, KP_REL = 2368;

  char* ws = (char*)d_ws;
  size_t cur = 0;
  auto alloc = [&](size_t bytes) -> void* {
    void* p = ws + cur;
    cur += (bytes + 255) & ~(size_t)255;
    return p;
  };
  float*     buf0 = (float*)alloc((size_t)B * 24 * 128 * 128 * 4); // conv l1/l3 out
  float*     buf1 = (float*)alloc((size_t)B * 24 * 64 * 64 * 4);   // conv l2/l4 out
  _Float16*  objF = (_Float16*)alloc((size_t)NB * KP_OBJ * 2);
  _Float16*  h0   = (_Float16*)alloc((size_t)NB * LAT * 2);
  _Float16*  h1   = (_Float16*)alloc((size_t)NB * LAT * 2);
  _Float16*  r0   = (_Float16*)alloc((size_t)NR * LAT * 2);
  _Float16*  r1   = (_Float16*)alloc((size_t)NR * LAT * 2);
  _Float16*  Wh0  = (_Float16*)alloc((size_t)LAT * KP_REL * 2);    // biggest first layer
  _Float16*  Wh1  = (_Float16*)alloc((size_t)LAT * LAT * 2);
  _Float16*  Wh2  = (_Float16*)alloc((size_t)LAT * LAT * 2);

  // ---- conv backbone (leaves 5..28: 4 x (W,b,g,be,mu,var)) ----
  {
    int Hin = 256, Cin = 3;
    const float* src = features;
    float* bufs[2] = {buf0, buf1};
    for (int l = 0; l < 4; ++l) {
      const float* Wc  = (const float*)d_in[5 + 6 * l + 0];
      const float* bc  = (const float*)d_in[5 + 6 * l + 1];
      const float* gc  = (const float*)d_in[5 + 6 * l + 2];
      const float* bec = (const float*)d_in[5 + 6 * l + 3];
      const float* muc = (const float*)d_in[5 + 6 * l + 4];
      const float* vac = (const float*)d_in[5 + 6 * l + 5];
      float* dst = bufs[l & 1];
      long long total = (long long)B * (Hin / 2) * (Hin / 2);
      int blocks = (int)((total + 255) / 256);
      conv_bn_relu<<<blocks, 256, 0, stream>>>(src, Wc, bc, gc, bec, muc, vac,
                                               dst, B, Cin, Hin, Hin);
      src = dst; Cin = 24; Hin >>= 1;
    }
  }
  const float* fm = buf1; // 64 x 24 x 16 x 16

  // ---- roi align -> obj f16 (1024 x 1184; 1176..1179 = pred bbox, 1180..1183 = 0)
  {
    int total = NB * 24 * 49;
    roi_align_obj<<<(total + 255) / 256, 256, 0, stream>>>(fm, orig_bboxes, box_bidx, objF, NB);
    obj_tail<<<(NB * 8 + 255) / 256, 256, 0, stream>>>(pred_bboxes, objF, NB);
  }

  const int gridObj = (NB / 128) * (LAT / 128);  // 64 blocks
  const int gridRel = (NR / 128) * (LAT / 128);  // 960 blocks

  // head leaf bases (sorted dict keys): color=29, mat=43, relate=57, shape=77, size=91
  auto run_obj_head = [&](int base, int dout, int ooff) {
    const float *W0 = (const float*)d_in[base + 0],  *b0 = (const float*)d_in[base + 1];
    const float *g0 = (const float*)d_in[base + 2],  *be0 = (const float*)d_in[base + 3];
    const float *mu0 = (const float*)d_in[base + 4], *va0 = (const float*)d_in[base + 5];
    const float *W1 = (const float*)d_in[base + 6],  *b1 = (const float*)d_in[base + 7];
    const float *g1 = (const float*)d_in[base + 8],  *be1 = (const float*)d_in[base + 9];
    const float *mu1 = (const float*)d_in[base + 10], *va1 = (const float*)d_in[base + 11];
    const float *Wo = (const float*)d_in[base + 12], *bo = (const float*)d_in[base + 13];

    long long t0 = (long long)LAT * KP_OBJ;
    wcvt_pad<<<(int)((t0 + 255) / 256), 256, 0, stream>>>(W0, Wh0, LAT, 1180, KP_OBJ);
    long long t1 = (long long)LAT * LAT;
    wcvt_pad<<<(int)((t1 + 255) / 256), 256, 0, stream>>>(W1, Wh1, LAT, LAT, LAT);

    gemm_wmma_bnrelu<false><<<gridObj, 256, 0, stream>>>(
        objF, nullptr, Wh0, b0, g0, be0, mu0, va0, h0, NB, KP_OBJ, LAT);
    gemm_wmma_bnrelu<false><<<gridObj, 256, 0, stream>>>(
        h0, nullptr, Wh1, b1, g1, be1, mu1, va1, h1, NB, LAT, LAT);
    head_out<<<(NB + 255) / 256, 256, 0, stream>>>(h1, Wo, bo, out + ooff, NB, LAT, dout, 0);
  };

  run_obj_head(77, 3, 0);       // shape_p  -> out[0 .. 3072)
  run_obj_head(29, 8, 3072);    // color_p  -> out[3072 .. 11264)
  run_obj_head(43, 2, 11264);   // mat_p    -> out[11264 .. 13312)
  run_obj_head(91, 2, 13312);   // size_p   -> out[13312 .. 15360)

  // ---- relate head (base 57): 3 hidden layers, sigmoid output ----
  {
    const int base = 57;
    const float *W0 = (const float*)d_in[base + 0],  *b0 = (const float*)d_in[base + 1];
    const float *g0 = (const float*)d_in[base + 2],  *be0 = (const float*)d_in[base + 3];
    const float *mu0 = (const float*)d_in[base + 4], *va0 = (const float*)d_in[base + 5];
    const float *W1 = (const float*)d_in[base + 6],  *b1 = (const float*)d_in[base + 7];
    const float *g1 = (const float*)d_in[base + 8],  *be1 = (const float*)d_in[base + 9];
    const float *mu1 = (const float*)d_in[base + 10], *va1 = (const float*)d_in[base + 11];
    const float *W2 = (const float*)d_in[base + 12], *b2 = (const float*)d_in[base + 13];
    const float *g2 = (const float*)d_in[base + 14], *be2 = (const float*)d_in[base + 15];
    const float *mu2 = (const float*)d_in[base + 16], *va2 = (const float*)d_in[base + 17];
    const float *Wo = (const float*)d_in[base + 18], *bo = (const float*)d_in[base + 19];

    long long t0 = (long long)LAT * KP_REL;
    wcvt_relate0<<<(int)((t0 + 255) / 256), 256, 0, stream>>>(W0, Wh0, LAT);
    long long t1 = (long long)LAT * LAT;
    wcvt_pad<<<(int)((t1 + 255) / 256), 256, 0, stream>>>(W1, Wh1, LAT, LAT, LAT);
    wcvt_pad<<<(int)((t1 + 255) / 256), 256, 0, stream>>>(W2, Wh2, LAT, LAT, LAT);

    // layer 0: inline pair-gather from objF (avoids 72 MB rela matrix)
    gemm_wmma_bnrelu<true><<<gridRel, 256, 0, stream>>>(
        objF, rela_pairs, Wh0, b0, g0, be0, mu0, va0, r0, NR, KP_REL, LAT);
    gemm_wmma_bnrelu<false><<<gridRel, 256, 0, stream>>>(
        r0, nullptr, Wh1, b1, g1, be1, mu1, va1, r1, NR, LAT, LAT);
    gemm_wmma_bnrelu<false><<<gridRel, 256, 0, stream>>>(
        r1, nullptr, Wh2, b2, g2, be2, mu2, va2, r0, NR, LAT, LAT);
    head_out<<<(NR + 255) / 256, 256, 0, stream>>>(r0, Wo, bo, out + 15360, NR, LAT, 2, 1);
  }
}
